// BaseUNet_31997506355335
// MI455X (gfx1250) — matrix operand order repaired and checked by reading, not compiled, not run on hardware
//
#include <hip/hip_runtime.h>

typedef __attribute__((ext_vector_type(16))) _Float16 v16h;
typedef __attribute__((ext_vector_type(8)))  _Float16 v8h;
typedef __attribute__((ext_vector_type(2)))  _Float16 v2h;
typedef __attribute__((ext_vector_type(8)))  float    v8f;

#define NNODES 50000
#define NEDGES 800000
#define NBATCH 16
#define BN_EPS 1e-5f

// ---------------------------------------------------------------- utility
__global__ void fill_kernel(float* __restrict__ p, float v, int n) {
  int i = blockIdx.x * blockDim.x + threadIdx.x;
  if (i < n) p[i] = v;
}

__global__ void cvt_f16_kernel(const float* __restrict__ w, _Float16* __restrict__ wh, int n) {
  int i = blockIdx.x * blockDim.x + threadIdx.x;
  if (i < n) wh[i] = (_Float16)w[i];
}

__global__ void deg_edges_kernel(const int* __restrict__ ei, float* __restrict__ deg, int E) {
  int e = blockIdx.x * blockDim.x + threadIdx.x;
  if (e < E) atomicAdd(&deg[ei[E + e]], 1.0f);
}

__global__ void rsqrt_ip_kernel(float* __restrict__ p, int n) {
  int i = blockIdx.x * blockDim.x + threadIdx.x;
  if (i < n) p[i] = rsqrtf(p[i]);
}

// ---------------------------------------------------------------- WMMA GEMM
// Y[N,M] = X[N,K] @ W[M,K]^T + bias, W pre-converted to f16 (row-major [M,K]).
// N must be a multiple of 16. Block = 256 threads (8 waves); block tile is
// 16 rows x 512 cols, each wave owns 64 cols (4 WMMA tiles). The 16x32 f16
// A-tile is staged in LDS once per k-step and shared by all 8 waves.
// KAL: K % 32 == 0 (unguarded vector loads); MAL: M % 16 == 0.
template <bool KAL, bool MAL>
__global__ void gemm_wmma_kernel(const float* __restrict__ X, const _Float16* __restrict__ Wh,
                                 const float* __restrict__ bias, float* __restrict__ Y,
                                 int N, int K, int M) {
  __shared__ _Float16 aT[16 * 32];
  const int lane = threadIdx.x & 31;
  const int wave = threadIdx.x >> 5;
  const int rb = blockIdx.x * 16;              // row base
  const int cb = blockIdx.y * 512 + wave * 64; // col base for this wave

  v8f acc[4] = {};

  const int mrow = lane & 15;
  const int khiA = (lane >= 16) ? 16 : 0;  // contiguous K order, matches B
  const int khiB = (lane >= 16) ? 16 : 0;  // contiguous K order, matches A

  // staging indices: thread t covers row t>>4, k-pair (t&15)*2
  const int sR = threadIdx.x >> 4;
  const int sK = (threadIdx.x & 15) << 1;

  for (int k0 = 0; k0 < K; k0 += 32) {
    __syncthreads();
    if (KAL) {
      const float2 xv = *(const float2*)(X + (size_t)(rb + sR) * K + k0 + sK);
      v2h p;
      p[0] = (_Float16)xv.x;
      p[1] = (_Float16)xv.y;
      *(v2h*)(aT + sR * 32 + sK) = p;
    } else {
      v2h p;
      p[0] = (k0 + sK     < K) ? (_Float16)X[(size_t)(rb + sR) * K + k0 + sK]     : (_Float16)0.0f;
      p[1] = (k0 + sK + 1 < K) ? (_Float16)X[(size_t)(rb + sR) * K + k0 + sK + 1] : (_Float16)0.0f;
      *(v2h*)(aT + sR * 32 + sK) = p;
    }
    __syncthreads();

    // A fragment: halves 0..15 are K = khiA..khiA+15.
    const v8h alo = *(const v8h*)(aT + mrow * 32 + khiA);
    const v8h ahi = *(const v8h*)(aT + mrow * 32 + khiA + 8);
    const v16h a = __builtin_shufflevector(alo, ahi, 0, 1, 2, 3, 4, 5, 6, 7,
                                           8, 9, 10, 11, 12, 13, 14, 15);

#pragma unroll
    for (int ct = 0; ct < 4; ++ct) {
      const int col = cb + ct * 16 + (lane & 15);
      v16h b;
      if (KAL) {
        if (MAL || col < M) {
          const _Float16* wp = Wh + (size_t)col * K + k0 + khiB;
          const v8h b0 = *(const v8h*)wp;
          const v8h b1 = *(const v8h*)(wp + 8);
          b = __builtin_shufflevector(b0, b1, 0, 1, 2, 3, 4, 5, 6, 7,
                                      8, 9, 10, 11, 12, 13, 14, 15);
        } else {
          b = (v16h)(_Float16)0.0f;
        }
      } else {
#pragma unroll
        for (int h = 0; h < 16; ++h) {
          const int gk = k0 + h + khiB;
          b[h] = ((MAL || col < M) && gk < K) ? Wh[(size_t)col * K + gk] : (_Float16)0.0f;
        }
      }
      acc[ct] = __builtin_amdgcn_wmma_f32_16x16x32_f16(false, a, false, b,
                                                       (short)0, acc[ct], false, false);
    }
  }

  const int rOff = (lane >= 16) ? 8 : 0;  // ISA 32-bit C/D 16x16 layout
#pragma unroll
  for (int ct = 0; ct < 4; ++ct) {
    const int col = cb + ct * 16 + (lane & 15);
    if (MAL || col < M) {
      const float bv = (bias != nullptr) ? bias[col] : 0.0f;
#pragma unroll
      for (int r = 0; r < 8; ++r) {
        Y[(size_t)(rb + r + rOff) * M + col] = acc[ct][r] + bv;
      }
    }
  }
}

// ---------------------------------------------------------------- BatchNorm
__global__ void bn_stats_kernel(const float* __restrict__ X, float* __restrict__ stats,
                                int n, int C) {
  int c = blockIdx.x * 64 + threadIdx.x;  // grid.x = C/64, block = 64
  int r0 = blockIdx.y * 512;
  int r1 = min(r0 + 512, n);
  float s = 0.0f, s2 = 0.0f;
  for (int r = r0; r < r1; ++r) {
    float v = X[(size_t)r * C + c];
    s += v; s2 += v * v;
  }
  atomicAdd(&stats[c], s);
  atomicAdd(&stats[C + c], s2);
}

__global__ void bn_apply_relu_kernel(float* __restrict__ h, const float* __restrict__ stats,
                                     const float* __restrict__ g, const float* __restrict__ b,
                                     long long total, int C, float invN) {
  long long i = (long long)blockIdx.x * blockDim.x + threadIdx.x;
  if (i >= total) return;
  int c = (int)(i & (C - 1));  // C is a power of two
  float m = stats[c] * invN;
  float v = stats[C + c] * invN - m * m;
  float y = (h[i] - m) * rsqrtf(v + BN_EPS) * g[c] + b[c];
  h[i] = fmaxf(y, 0.0f);
}

// ---------------------------------------------------------------- GCN scatter
// self-loop term fused: out = bias + dis[n]^2 * h
__global__ void scatter_init_kernel(const float* __restrict__ h, const float* __restrict__ dis,
                                    const float* __restrict__ bias, float* __restrict__ out,
                                    long long total, int C, int cshift) {
  long long i = (long long)blockIdx.x * blockDim.x + threadIdx.x;
  if (i >= total) return;
  int node = (int)(i >> cshift);
  int c = (int)(i & (C - 1));
  float d = dis[node];
  out[i] = bias[c] + d * d * h[i];
}

__global__ void scatter_edges_kernel(const float* __restrict__ h, const float* __restrict__ dis,
                                     const int* __restrict__ ei, float* __restrict__ out,
                                     int E, int C, int chunkShift, int total) {
  int t = blockIdx.x * blockDim.x + threadIdx.x;
  if (t >= total) return;
  int e = t >> chunkShift;
  int c4 = (t & ((1 << chunkShift) - 1)) << 2;
  int s = ei[e], d = ei[E + e];
  float coef = dis[s] * dis[d];
  const float4 hv = *(const float4*)(h + (size_t)s * C + c4);
  float* op = out + (size_t)d * C + c4;
  atomicAdd(op + 0, coef * hv.x);
  atomicAdd(op + 1, coef * hv.y);
  atomicAdd(op + 2, coef * hv.z);
  atomicAdd(op + 3, coef * hv.w);
}

// ---------------------------------------------------------------- pooling
__global__ void pool_add_kernel(const float* __restrict__ h, const int* __restrict__ batch,
                                float* __restrict__ g, float* __restrict__ cnt,
                                long long total, int C, int cshift) {
  long long i = (long long)blockIdx.x * blockDim.x + threadIdx.x;
  if (i >= total) return;
  int n = (int)(i >> cshift);
  int c = (int)(i & (C - 1));
  int b = batch[n];
  atomicAdd(&g[(size_t)b * C + c], h[i]);
  if (c == 0) atomicAdd(&cnt[b], 1.0f);
}

__global__ void pool_div_kernel(float* __restrict__ g, const float* __restrict__ cnt,
                                int total, int cshift) {
  int i = blockIdx.x * blockDim.x + threadIdx.x;
  if (i >= total) return;
  g[i] /= fmaxf(cnt[i >> cshift], 1.0f);
}

// ---------------------------------------------------------------- host
extern "C" void kernel_launch(void* const* d_in, const int* in_sizes, int n_in,
                              void* d_out, int out_size, void* d_ws, size_t ws_size,
                              hipStream_t stream) {
  (void)in_sizes; (void)n_in; (void)out_size; (void)ws_size;
  const int N = NNODES, E = NEDGES, B = NBATCH;

  const float* x = (const float*)d_in[0];
  const int* ei = (const int*)d_in[1];
  const int* batch = (const int*)d_in[2];
  auto F = [&](int i) { return (const float*)d_in[i]; };

  // workspace carve-out (256B aligned)
  char* ws = (char*)d_ws;
  size_t off = 0;
  auto alloc = [&](size_t bytes) -> void* {
    char* p = ws + off;
    off = (off + bytes + 255) & ~(size_t)255;
    return (void*)p;
  };
  float*    dis   = (float*)alloc((size_t)N * 4);
  float*    bufA  = (float*)alloc((size_t)N * 512 * 4);
  float*    bufB  = (float*)alloc((size_t)N * 512 * 4);
  float*    stats = (float*)alloc(2 * 2048 * 4);
  float*    gA    = (float*)alloc((size_t)B * 2048 * 4);
  float*    gB    = (float*)alloc((size_t)B * 2048 * 4);
  float*    cnt   = (float*)alloc((size_t)B * 4);
  _Float16* whBuf = (_Float16*)alloc((size_t)3750 * 2048 * 2);  // largest weight

  dim3 blk(256);
  auto grid1 = [](long long n, int b) { return dim3((unsigned)((n + b - 1) / b)); };

  // symmetric normalization: deg (with self loop) -> dis = rsqrt(deg)
  fill_kernel<<<grid1(N, 256), blk, 0, stream>>>(dis, 1.0f, N);
  deg_edges_kernel<<<grid1(E, 256), blk, 0, stream>>>(ei, dis, E);
  rsqrt_ip_kernel<<<grid1(N, 256), blk, 0, stream>>>(dis, N);

  // GEMM wrapper: converts W -> f16 workspace, then WMMA GEMM
  auto gemm = [&](const float* X, const float* W, const float* bias, float* Y,
                  int n, int k, int m) {
    cvt_f16_kernel<<<grid1((long long)m * k, 256), blk, 0, stream>>>(W, whBuf, m * k);
    dim3 g(n / 16, (m + 511) / 512);
    if ((k & 31) == 0 && (m & 15) == 0)
      gemm_wmma_kernel<true, true><<<g, dim3(256), 0, stream>>>(X, whBuf, bias, Y, n, k, m);
    else if ((k & 31) == 0)
      gemm_wmma_kernel<true, false><<<g, dim3(256), 0, stream>>>(X, whBuf, bias, Y, n, k, m);
    else
      gemm_wmma_kernel<false, true><<<g, dim3(256), 0, stream>>>(X, whBuf, bias, Y, n, k, m);
  };
  auto bn = [&](float* h, const float* gam, const float* bet, int n, int C) {
    fill_kernel<<<grid1(2 * C, 256), blk, 0, stream>>>(stats, 0.0f, 2 * C);
    dim3 sg(C / 64, (unsigned)((n + 511) / 512));
    bn_stats_kernel<<<sg, dim3(64), 0, stream>>>(h, stats, n, C);
    long long tot = (long long)n * C;
    bn_apply_relu_kernel<<<grid1(tot, 256), blk, 0, stream>>>(
        h, stats, gam, bet, tot, C, 1.0f / (float)n);
  };

  const int dims[5] = {64, 64, 128, 256, 512};
  const float* cur = x;
  int K = 5;
  for (int l = 0; l < 5; ++l) {
    int C = dims[l];
    const float* mlpW  = F(3 + l * 8 + 0);
    const float* mlpb  = F(3 + l * 8 + 1);
    const float* bn1g  = F(3 + l * 8 + 2);
    const float* bn1b  = F(3 + l * 8 + 3);
    const float* convW = F(3 + l * 8 + 4);
    const float* convb = F(3 + l * 8 + 5);
    const float* bn2g  = F(3 + l * 8 + 6);
    const float* bn2b  = F(3 + l * 8 + 7);

    float* t1 = (cur == bufA) ? bufB : bufA;  // scratch != cur
    float* t2 = (t1 == bufA) ? bufB : bufA;   // other buffer (may alias cur)

    gemm(cur, mlpW, mlpb, t1, N, K, C);       // h = X @ mlp_W^T + b
    bn(t1, bn1g, bn1b, N, C);                 // BN1 + ReLU (in place)
    gemm(t1, convW, nullptr, t2, N, C, C);    // h = h @ conv_W^T
    long long totNC = (long long)N * C;
    int cshift = __builtin_ctz(C);
    scatter_init_kernel<<<grid1(totNC, 256), blk, 0, stream>>>(
        t2, dis, convb, t1, totNC, C, cshift);  // bias + self-loop term
    int chunks = C >> 2;
    int chsh = __builtin_ctz(chunks);
    long long etot = (long long)E * chunks;
    scatter_edges_kernel<<<grid1(etot, 256), blk, 0, stream>>>(
        t2, dis, ei, t1, E, C, chsh, (int)etot);
    bn(t1, bn2g, bn2b, N, C);                 // BN2 + ReLU
    cur = t1;
    K = C;
  }

  // global mean pool -> gA [B,512]
  fill_kernel<<<grid1(B * 512, 256), blk, 0, stream>>>(gA, 0.0f, B * 512);
  fill_kernel<<<grid1(B, 256), blk, 0, stream>>>(cnt, 0.0f, B);
  pool_add_kernel<<<grid1((long long)N * 512, 256), blk, 0, stream>>>(
      cur, batch, gA, cnt, (long long)N * 512, 512, 9);
  pool_div_kernel<<<grid1(B * 512, 256), blk, 0, stream>>>(gA, cnt, B * 512, 9);

  // decoder: dec2 (512->1024)+BN+ReLU, dec1 (1024->2048)+BN+ReLU, dec0 (2048->3750)
  gemm(gA, F(43), F(44), gB, B, 512, 1024);
  bn(gB, F(45), F(46), B, 1024);
  gemm(gB, F(47), F(48), gA, B, 1024, 2048);
  bn(gA, F(49), F(50), B, 2048);
  gemm(gA, F(51), F(52), (float*)d_out, B, 2048, 3750);
}
